// TransductiveLearner_17867063951658
// MI455X (gfx1250) — matrix operand order repaired and checked by reading, not compiled, not run on hardware
//
#include <hip/hip_runtime.h>
#include <hip/hip_bf16.h>
#include <math.h>

// ---------------- problem constants (from reference) ----------------
constexpr int kNS = 1024;    // support rows
constexpr int kNT = 7168;    // test rows
constexpr int kN  = 8192;    // total rows (concat)
constexpr int kD  = 512;     // embedding dim
constexpr int kC  = 100;     // classes
constexpr int kCP = 112;     // classes padded to multiple of 16

// ---------------- workspace layout (bytes) ----------------
constexpr size_t OFF_ACC = 0;                                   // 8 floats (accumulators)
constexpr size_t OFF_SQ  = 1024;                                // kN floats (row sq-norms)
constexpr size_t OFF_EHI = 65536;                               // kN*kD bf16 (hi split)
constexpr size_t OFF_ELO = OFF_EHI + (size_t)kN * kD * 2;       // kN*kD bf16 (lo split)
constexpr size_t OFF_WHI = OFF_ELO + (size_t)kN * kD * 2;       // kCP*kD bf16 (W^T hi, padded)
constexpr size_t OFF_WLO = OFF_WHI + (size_t)kCP * kD * 2;      // kCP*kD bf16 (W^T lo)

// ---------------- smoothness kernel tiling ----------------
constexpr int kMR   = 64;    // rows per workgroup (4 row-subtiles x 16)
constexpr int kAStr = 520;   // padded A row stride in elems (bank-conflict-free)
constexpr int kBK   = 128;   // staged K panel depth
constexpr int kBStr = 136;   // padded B row stride in elems (bank-conflict-free)
constexpr int kNBuf = 3;     // triple-buffered B panels (async pipeline)
constexpr int kBufElems = 2 * 32 * kBStr;                 // hi+lo panel, elems
constexpr int kPanels = (kN / 32) * (kD / kBK);           // 1024

constexpr size_t kSmoothLds =
    (size_t)(2 * kMR * kAStr) * 2 +                       // A hi/lo
    (size_t)(kNBuf * kBufElems) * 2 +                     // B ring
    (size_t)8 * 16 * 3 * 4;                               // reduce scratch

// ---------------- WMMA fragment types ----------------
typedef __attribute__((ext_vector_type(16))) __bf16 bf16x16;
typedef __attribute__((ext_vector_type(8)))  float  v8f;

union FragBF { bf16x16 f; uint4 q[2]; };

__device__ __forceinline__ unsigned short f2bf_rne(float x) {
  unsigned int u = __float_as_uint(x);
  unsigned int r = u + 0x7fffu + ((u >> 16) & 1u);
  return (unsigned short)(r >> 16);
}
__device__ __forceinline__ float bf2f(unsigned short h) {
  return __uint_as_float(((unsigned int)h) << 16);
}

// ---- CDNA5 async global->LDS copy (ASYNCcnt path, see CDNA5 ISA) ----
__device__ __forceinline__ void async_b128_to_lds(void* lds_ptr, const void* gptr) {
#if defined(__gfx1250__)
  unsigned int lo = (unsigned int)(uintptr_t)lds_ptr;     // wave-LDS-relative offset
  unsigned long long ga = (unsigned long long)(uintptr_t)gptr;
  asm volatile("global_load_async_to_lds_b128 %0, %1, off"
               :: "v"(lo), "v"(ga) : "memory");
#else
  // host pass / non-gfx1250: synchronous fallback
  *(uint4*)lds_ptr = *(const uint4*)gptr;
#endif
}
__device__ __forceinline__ void async_wait_le(int n) {
#if defined(__gfx1250__)
  if (n == 0)
    asm volatile("s_wait_asynccnt 0x0" ::: "memory");
  else
    asm volatile("s_wait_asynccnt 0x4" ::: "memory");
#else
  (void)n;
#endif
}

// =====================================================================
// Kernel 1: split embeddings into bf16 hi/lo, compute exact fp32 row
// squared norms, and zero the loss accumulators.
// grid = kN blocks x 256 threads
// =====================================================================
__global__ __launch_bounds__(256) void k_prep_embed(
    const float* __restrict__ sup, const float* __restrict__ tst,
    unsigned short* __restrict__ ehi, unsigned short* __restrict__ elo,
    float* __restrict__ sq, float* __restrict__ acc) {
  int row = blockIdx.x;
  int tid = threadIdx.x;
  const float* src = (row < kNS) ? (sup + (size_t)row * kD)
                                 : (tst + (size_t)(row - kNS) * kD);
  float ss = 0.f;
  for (int e = tid; e < kD; e += 256) {
    float x = src[e];
    unsigned short h = f2bf_rne(x);
    ehi[(size_t)row * kD + e] = h;
    elo[(size_t)row * kD + e] = f2bf_rne(x - bf2f(h));
    ss += x * x;
  }
  for (int o = 16; o > 0; o >>= 1) ss += __shfl_down(ss, o, 32);
  __shared__ float wred[8];
  int wave = tid >> 5, lane = tid & 31;
  if (lane == 0) wred[wave] = ss;
  __syncthreads();
  if (tid == 0) {
    float t = 0.f;
    for (int w = 0; w < 8; ++w) t += wred[w];
    sq[row] = t;
  }
  if (row == 0 && tid < 8) acc[tid] = 0.f;
}

// =====================================================================
// Kernel 2: W [kD x kC] row-major -> transposed, padded bf16 hi/lo
// [kCP x kD] so B-fragments are contiguous in K.  grid = kCP blocks.
// =====================================================================
__global__ __launch_bounds__(256) void k_prep_w(
    const float* __restrict__ W,
    unsigned short* __restrict__ whi, unsigned short* __restrict__ wlo) {
  int c = blockIdx.x;
  for (int k = threadIdx.x; k < kD; k += 256) {
    float x = (c < kC) ? W[(size_t)k * kC + c] : 0.f;
    unsigned short h = f2bf_rne(x);
    whi[(size_t)c * kD + k] = h;
    wlo[(size_t)c * kD + k] = f2bf_rne(x - bf2f(h));
  }
}

// =====================================================================
// Kernel 3: logits = E @ W + b via bf16x3 WMMA, then per-row logsumexp
// epilogue -> support CE (acc[0]) and test KL (acc[1]).
// grid = (kNS+kNT)/16 blocks x 256 threads (waves 0..6 own col tiles)
// =====================================================================
__global__ __launch_bounds__(256) void k_logits_loss(
    const unsigned short* __restrict__ ehi, const unsigned short* __restrict__ elo,
    const unsigned short* __restrict__ whi, const unsigned short* __restrict__ wlo,
    const float* __restrict__ b, const float* __restrict__ preds,
    const int* __restrict__ sy, float* __restrict__ acc) {
  __shared__ float Lg[16][kCP];
  __shared__ float bsh[kCP];
  int tid = threadIdx.x;
  int blk = blockIdx.x;
  bool is_sup = blk < (kNS / 16);
  int row0_local = is_sup ? blk * 16 : (blk - kNS / 16) * 16;  // row within its set
  int erow0 = is_sup ? row0_local : kNS + row0_local;          // row within concat E

  if (tid < kCP) bsh[tid] = (tid < kC) ? b[tid] : 0.f;

  int wave = tid >> 5, lane = tid & 31;
  int lane_m = lane & 15, hi_half = lane >> 4;

  if (wave < 7) {
    int cbase = wave * 16;
    int kgroup = hi_half * 8;
    v8f accv = {};
    const unsigned short* arow_hi = ehi + (size_t)(erow0 + lane_m) * kD;
    const unsigned short* arow_lo = elo + (size_t)(erow0 + lane_m) * kD;
    const unsigned short* bcol_hi = whi + (size_t)(cbase + lane_m) * kD;
    const unsigned short* bcol_lo = wlo + (size_t)(cbase + lane_m) * kD;
    for (int ks = 0; ks < kD; ks += 32) {
      FragBF ah, al, bh, bl;
      ah.q[0] = *(const uint4*)(arow_hi + ks + kgroup);
      ah.q[1] = *(const uint4*)(arow_hi + ks + 16 + kgroup);
      al.q[0] = *(const uint4*)(arow_lo + ks + kgroup);
      al.q[1] = *(const uint4*)(arow_lo + ks + 16 + kgroup);
      bh.q[0] = *(const uint4*)(bcol_hi + ks + kgroup);
      bh.q[1] = *(const uint4*)(bcol_hi + ks + 16 + kgroup);
      bl.q[0] = *(const uint4*)(bcol_lo + ks + kgroup);
      bl.q[1] = *(const uint4*)(bcol_lo + ks + 16 + kgroup);
      accv = __builtin_amdgcn_wmma_f32_16x16x32_bf16(false, ah.f, false, bh.f, (short)0, accv, false, false);
      accv = __builtin_amdgcn_wmma_f32_16x16x32_bf16(false, ah.f, false, bl.f, (short)0, accv, false, false);
      accv = __builtin_amdgcn_wmma_f32_16x16x32_bf16(false, al.f, false, bh.f, (short)0, accv, false, false);
    }
#pragma unroll
    for (int r = 0; r < 8; ++r)
      Lg[hi_half ? 8 + r : r][cbase + lane_m] = accv[r];
  }
  __syncthreads();

  if (tid < 16) {
    int row = row0_local + tid;
    float mx = -1e30f;
    for (int c = 0; c < kC; ++c) mx = fmaxf(mx, Lg[tid][c] + bsh[c]);
    float s = 0.f;
    for (int c = 0; c < kC; ++c) s += __expf(Lg[tid][c] + bsh[c] - mx);
    float lse = __logf(s) + mx;
    if (is_sup) {
      int y = sy[row];
      if (y < 0) y = 0;
      if (y >= kC) y = kC - 1;
      atomicAdd(&acc[0], lse - (Lg[tid][y] + bsh[y]));
    } else {
      float kl = 0.f;
      const float* p = preds + (size_t)row * kC;
      for (int c = 0; c < kC; ++c) {
        float pv = p[c];
        if (pv > 0.f) kl += pv * (__logf(pv) - (Lg[tid][c] + bsh[c] - lse));
      }
      atomicAdd(&acc[1], kl);
    }
  }
}

// =====================================================================
// Kernel 4: streaming smoothness.  Gram tiles via bf16x3 WMMA, fused
// online-softmax over columns (flash style), dist from Gram identity.
// B panels are triple-buffered and streamed into LDS with CDNA5 async
// global->LDS copies (ASYNCcnt), overlapping L2 latency with WMMA.
// grid = kN/kMR blocks x 256 threads, dynamic LDS = kSmoothLds bytes.
// 8 waves = 4 row-subtiles x 2 col-subtiles.
// =====================================================================
__global__ __launch_bounds__(256) void k_smooth(
    const unsigned short* __restrict__ ehi, const unsigned short* __restrict__ elo,
    const float* __restrict__ sq, float* __restrict__ acc) {
  extern __shared__ char smem[];
  unsigned short* Ahi = (unsigned short*)smem;
  unsigned short* Alo = Ahi + kMR * kAStr;
  unsigned short* Bring = Alo + kMR * kAStr;                  // kNBuf panels (hi+lo each)
  float* red = (float*)(Bring + kNBuf * kBufElems);

  int tid = threadIdx.x;
  int row_base = blockIdx.x * kMR;

  // --- stage A (kMR rows x kD, hi & lo) into LDS, coalesced as dwords ---
  {
    const unsigned int* shi = (const unsigned int*)ehi + (size_t)row_base * (kD / 2);
    const unsigned int* slo = (const unsigned int*)elo + (size_t)row_base * (kD / 2);
    unsigned int* dhi = (unsigned int*)Ahi;
    unsigned int* dlo = (unsigned int*)Alo;
    for (int idx = tid; idx < kMR * (kD / 2); idx += 256) {
      int r = idx >> 8;        // kD/2 = 256 dwords per row
      int kk = idx & 255;
      dhi[r * (kAStr / 2) + kk] = shi[(size_t)r * (kD / 2) + kk];
      dlo[r * (kAStr / 2) + kk] = slo[(size_t)r * (kD / 2) + kk];
    }
  }
  __syncthreads();

  int wave = tid >> 5, lane = tid & 31;
  int rg = wave >> 1, cg = wave & 1;
  int lane_m = lane & 15, hi_half = lane >> 4;
  int kgroup = hi_half * 8;
  int row16 = rg * 16;

  // exact fp32 row norms for my 8 rows (fixed for the whole kernel)
  float sqi[8];
#pragma unroll
  for (int r = 0; r < 8; ++r)
    sqi[r] = sq[row_base + row16 + (hi_half ? 8 + r : r)];

  float m_run[8], d_run[8], s_run[8];
#pragma unroll
  for (int r = 0; r < 8; ++r) { m_run[r] = -1e30f; d_run[r] = 0.f; s_run[r] = 0.f; }

  const unsigned short* Arow_hi = Ahi + (row16 + lane_m) * kAStr;
  const unsigned short* Arow_lo = Alo + (row16 + lane_m) * kAStr;

  // issue the 4 per-thread async B128 copies for panel p into ring slot p%3
  auto issue_panel = [&](int p) {
    int jb = (p >> 2) * 32;
    int ko = (p & 3) * kBK;
    unsigned short* bh = Bring + (p % kNBuf) * kBufElems;
    unsigned short* bl = bh + 32 * kBStr;
#pragma unroll
    for (int it = 0; it < 2; ++it) {
      int chunk = tid + it * 256;       // 512 B128 chunks per matrix
      int r = chunk >> 4;               // panel row (gram column)
      int cc = chunk & 15;              // 16B chunk within the 256B row
      size_t goff = (size_t)(jb + r) * kD + ko + cc * 8;
      async_b128_to_lds(bh + r * kBStr + cc * 8, ehi + goff);
      async_b128_to_lds(bl + r * kBStr + cc * 8, elo + goff);
    }
  };

  issue_panel(0);
  v8f accv = {};

  for (int p = 0; p < kPanels; ++p) {
    if (p + 1 < kPanels) {
      issue_panel(p + 1);
      async_wait_le(4);                 // panel p's 4 per-thread copies done
    } else {
      async_wait_le(0);
    }
    __syncthreads();                    // whole panel p visible to all waves

    if ((p & 3) == 0) {
#pragma unroll
      for (int r = 0; r < 8; ++r) accv[r] = 0.f;
    }
    int ko = (p & 3) * kBK;
    const unsigned short* bh = Bring + (p % kNBuf) * kBufElems;
    const unsigned short* bl = bh + 32 * kBStr;
    const unsigned short* Brow_hi = bh + (cg * 16 + lane_m) * kBStr + kgroup;
    const unsigned short* Brow_lo = bl + (cg * 16 + lane_m) * kBStr + kgroup;

#pragma unroll
    for (int ki = 0; ki < kBK; ki += 32) {
      int ks = ko + ki;
      FragBF ah, al, fbh, fbl;
      ah.q[0] = *(const uint4*)(Arow_hi + ks + kgroup);
      ah.q[1] = *(const uint4*)(Arow_hi + ks + 16 + kgroup);
      al.q[0] = *(const uint4*)(Arow_lo + ks + kgroup);
      al.q[1] = *(const uint4*)(Arow_lo + ks + 16 + kgroup);
      fbh.q[0] = *(const uint4*)(Brow_hi + ki);
      fbh.q[1] = *(const uint4*)(Brow_hi + ki + 16);
      fbl.q[0] = *(const uint4*)(Brow_lo + ki);
      fbl.q[1] = *(const uint4*)(Brow_lo + ki + 16);
      accv = __builtin_amdgcn_wmma_f32_16x16x32_bf16(false, ah.f, false, fbh.f, (short)0, accv, false, false);
      accv = __builtin_amdgcn_wmma_f32_16x16x32_bf16(false, ah.f, false, fbl.f, (short)0, accv, false, false);
      accv = __builtin_amdgcn_wmma_f32_16x16x32_bf16(false, al.f, false, fbh.f, (short)0, accv, false, false);
    }

    if ((p & 3) == 3) {
      // --- online-softmax epilogue for this 16-col slab ---
      int jb = (p >> 2) * 32;
      int j = jb + cg * 16 + lane_m;
      float sqj = sq[j];
#pragma unroll
      for (int r = 0; r < 8; ++r) {
        float g = accv[r];
        float l = g * 10.0f;  // / SMOOTH_TEMP
        float mx = l;
        for (int o = 1; o < 16; o <<= 1) mx = fmaxf(mx, __shfl_xor(mx, o, 16));
        float newm = fmaxf(m_run[r], mx);
        float e = __expf(l - newm);
        float d2 = sqi[r] + sqj - 2.0f * g;
        float dist = (d2 > 0.f) ? sqrtf(d2) : 0.f;
        float es = e, eds = e * dist;
        for (int o = 1; o < 16; o <<= 1) {
          es += __shfl_xor(es, o, 16);
          eds += __shfl_xor(eds, o, 16);
        }
        float scale = __expf(m_run[r] - newm);
        d_run[r] = d_run[r] * scale + es;
        s_run[r] = s_run[r] * scale + eds;
        m_run[r] = newm;
      }
    }
  }

  // --- merge the two column-waves per row, then accumulate ---
  __syncthreads();
  if (lane == 0 || lane == 16) {
#pragma unroll
    for (int r = 0; r < 8; ++r) {
      int lr = hi_half ? 8 + r : r;
      int base = (wave * 16 + lr) * 3;
      red[base + 0] = m_run[r];
      red[base + 1] = d_run[r];
      red[base + 2] = s_run[r];
    }
  }
  __syncthreads();
  if (tid < kMR) {
    int rg2 = tid >> 4, lr = tid & 15;
    int w0 = rg2 * 2, w1 = w0 + 1;
    float m0 = red[(w0 * 16 + lr) * 3 + 0], d0 = red[(w0 * 16 + lr) * 3 + 1], s0 = red[(w0 * 16 + lr) * 3 + 2];
    float m1 = red[(w1 * 16 + lr) * 3 + 0], d1 = red[(w1 * 16 + lr) * 3 + 1], s1 = red[(w1 * 16 + lr) * 3 + 2];
    float M = fmaxf(m0, m1);
    float e0 = __expf(m0 - M), e1 = __expf(m1 - M);
    float Dn = d0 * e0 + d1 * e1;
    float Sn = s0 * e0 + s1 * e1;
    atomicAdd(&acc[2], Sn / Dn);
  }
}

// =====================================================================
// Kernel 5: combine weighted terms into the scalar loss.
// =====================================================================
__global__ void k_final(const float* __restrict__ acc, float* __restrict__ out) {
  out[0] = acc[0] / (float)kNS + 0.1f * (acc[1] / (float)kNT) +
           0.01f * (acc[2] / (float)kN);
}

// =====================================================================
extern "C" void kernel_launch(void* const* d_in, const int* in_sizes, int n_in,
                              void* d_out, int out_size, void* d_ws, size_t ws_size,
                              hipStream_t stream) {
  (void)in_sizes; (void)n_in; (void)out_size; (void)ws_size;
  const float* sup   = (const float*)d_in[0];
  const float* tst   = (const float*)d_in[1];
  const float* preds = (const float*)d_in[2];
  const float* W     = (const float*)d_in[3];
  const float* b     = (const float*)d_in[4];
  const int*   sy    = (const int*)d_in[5];

  char* ws = (char*)d_ws;
  float* acc = (float*)(ws + OFF_ACC);
  float* sq  = (float*)(ws + OFF_SQ);
  unsigned short* ehi = (unsigned short*)(ws + OFF_EHI);
  unsigned short* elo = (unsigned short*)(ws + OFF_ELO);
  unsigned short* whi = (unsigned short*)(ws + OFF_WHI);
  unsigned short* wlo = (unsigned short*)(ws + OFF_WLO);

  k_prep_embed<<<kN, 256, 0, stream>>>(sup, tst, ehi, elo, sq, acc);
  k_prep_w<<<kCP, 256, 0, stream>>>(W, whi, wlo);
  k_logits_loss<<<(kNS + kNT) / 16, 256, 0, stream>>>(ehi, elo, whi, wlo, b, preds, sy, acc);
  k_smooth<<<kN / kMR, 256, kSmoothLds, stream>>>(ehi, elo, sq, acc);
  k_final<<<1, 1, 0, stream>>>(acc, (float*)d_out);
}
